// PredictorCrossAttention_61211873903138
// MI455X (gfx1250) — matrix operand order repaired and checked by reading, not compiled, not run on hardware
//
#include <hip/hip_runtime.h>
#include <hip/hip_bf16.h>

typedef float v2f  __attribute__((ext_vector_type(2)));
typedef float v8f  __attribute__((ext_vector_type(8)));
typedef __bf16 v16bf __attribute__((ext_vector_type(16)));
typedef __bf16 v8bf  __attribute__((ext_vector_type(8)));

union BFPack { v16bf v; v8bf h[2]; };

#define BD   8
#define TT   4096
#define DD   768
#define NH   12
#define HD   64
#define MM   64        // n_state
#define TC   256       // tokens per chunk
#define THF  128       // tokens per half-pass
#define NC   (TT / TC) // 16 chunks
#define KB   32        // K-block (d) for score GEMM
#define KPAD 40        // bf16 row stride (16B-aligned, bank-staggered)
#define WPAD (THF + 2) // Wbuf row stride in floats (8B-aligned rows)
#define ALPHA_C 0.1f

// ---------------- Kernel 0: KE[h][m][d] = sum_e proto[m][h*64+e] * W_tok[h*64+e][d]
__global__ __launch_bounds__(256) void ke_precompute_kernel(
    const float* __restrict__ proto, const float* __restrict__ Wt,
    float* __restrict__ KE) {
  int idx = blockIdx.x * 256 + threadIdx.x;
  if (idx >= NH * MM * DD) return;
  int d = idx % DD;
  int m = (idx / DD) % MM;
  int h = idx / (DD * MM);
  const float* pr = proto + (size_t)m * DD + h * HD;
  const float* wc = Wt + (size_t)(h * HD) * DD + d;
  float s = 0.f;
#pragma unroll 8
  for (int e = 0; e < HD; ++e) s += pr[e] * wc[(size_t)e * DD];
  KE[idx] = s;
}

// ---------------- Kernel 1: per (b, h, chunk): score GEMM (bf16x3 split-WMMA)
// -> softmax -> backward scan -> partial s GEMM (f32 WMMA).
__global__ __launch_bounds__(256) void score_scan_kernel(
    const float* __restrict__ H, const float* __restrict__ KE,
    float* __restrict__ s_chunks,  // [B][NH][NC][MM][HD]
    float* __restrict__ G_chunks)  // [B][NH][NC][MM]
{
  __shared__ __bf16 Hs2[2][THF][KPAD];  // hi/lo planes of H K-block  (20.0 KB)
  __shared__ __bf16 KEs2[2][MM][KPAD];  // hi/lo planes of KE K-block (10.0 KB)
  __shared__ float  Wbuf[MM][WPAD];     // scores -> w -> coeff, [m][t] (32.5 KB)
  __shared__ float  Pbuf[MM];

  const int blk = blockIdx.x;
  const int c = blk % NC;
  const int hh = (blk / NC) % NH;
  const int b = blk / (NC * NH);
  const int t_chunk = c * TC;

  const int tid  = threadIdx.x;
  const int lane = tid & 31;
  const int wave = tid >> 5;
  const int l15  = lane & 15;
  const int row  = (wave << 4) + l15;     // this wave's 16-row strip (0..127)
  const int kbA  = (lane >> 4) << 3;      // 16-bit A: 0 or 8
  const int kbB  = (lane >> 4) << 4;      // 16-bit B: 0 or 16
  const int koff = (lane >> 4) << 1;      // fp32 WMMA (2nd GEMM): 0 or 2

  const float* Hb  = H + (size_t)b * TT * DD;
  const float* KEh = KE + (size_t)hh * MM * DD;

  if (tid < MM) Pbuf[tid] = 1.0f;

  // persistent accumulators for the second GEMM (s tiles), carried across halves
  const int mt2 = wave & 3;
  const int et0 = (wave >> 2) << 1;
  v8f sacc0 = {}; v8f sacc1 = {};

  for (int half = 1; half >= 0; --half) {   // backward over halves (suffix scan)
    const int t0 = t_chunk + half * THF;

    // ---------- score GEMM: [128 x 768] x [768 x 64], bf16 hi/lo split ----------
    v8f acc[4] = {};
    for (int kb = 0; kb < DD / KB; ++kb) {
      __syncthreads();
      { // stage H K-block: 128 x 32 f32 -> hi/lo bf16
        int r = tid >> 1, cc = (tid & 1) * 16;
        const float4* src = (const float4*)(Hb + (size_t)(t0 + r) * DD + kb * KB + cc);
        float x[16];
        *(float4*)&x[0] = src[0]; *(float4*)&x[4] = src[1];
        *(float4*)&x[8] = src[2]; *(float4*)&x[12] = src[3];
        v8bf hi[2], lo[2];
#pragma unroll
        for (int i = 0; i < 16; ++i) {
          __bf16 h16 = (__bf16)x[i];
          hi[i >> 3][i & 7] = h16;
          lo[i >> 3][i & 7] = (__bf16)(x[i] - (float)h16);
        }
        *(v8bf*)&Hs2[0][r][cc]     = hi[0];
        *(v8bf*)&Hs2[0][r][cc + 8] = hi[1];
        *(v8bf*)&Hs2[1][r][cc]     = lo[0];
        *(v8bf*)&Hs2[1][r][cc + 8] = lo[1];
        if (kb + 1 < DD / KB)
          __builtin_prefetch(Hb + (size_t)(t0 + r) * DD + (kb + 1) * KB + cc, 0, 1);
      }
      { // stage KE K-block: 64 x 32 f32 -> hi/lo bf16
        int m = tid >> 2, cc = (tid & 3) * 8;
        const float4* src = (const float4*)(KEh + (size_t)m * DD + kb * KB + cc);
        float x[8];
        *(float4*)&x[0] = src[0]; *(float4*)&x[4] = src[1];
        v8bf hi, lo;
#pragma unroll
        for (int i = 0; i < 8; ++i) {
          __bf16 h16 = (__bf16)x[i];
          hi[i] = h16;
          lo[i] = (__bf16)(x[i] - (float)h16);
        }
        *(v8bf*)&KEs2[0][m][cc] = hi;
        *(v8bf*)&KEs2[1][m][cc] = lo;
      }
      __syncthreads();

      BFPack ahi, alo;
      ahi.h[0] = *(const v8bf*)&Hs2[0][row][kbA];
      ahi.h[1] = *(const v8bf*)&Hs2[0][row][kbA + 16];
      alo.h[0] = *(const v8bf*)&Hs2[1][row][kbA];
      alo.h[1] = *(const v8bf*)&Hs2[1][row][kbA + 16];
#pragma unroll
      for (int mt = 0; mt < 4; ++mt) {
        int mcol = (mt << 4) + l15;
        BFPack bhi, blo;
        bhi.h[0] = *(const v8bf*)&KEs2[0][mcol][kbA];
        bhi.h[1] = *(const v8bf*)&KEs2[0][mcol][kbA + 16];
        blo.h[0] = *(const v8bf*)&KEs2[1][mcol][kbA];
        blo.h[1] = *(const v8bf*)&KEs2[1][mcol][kbA + 16];
        acc[mt] = __builtin_amdgcn_wmma_f32_16x16x32_bf16(
            false, ahi.v, false, bhi.v, (short)0, acc[mt], false, false);
        acc[mt] = __builtin_amdgcn_wmma_f32_16x16x32_bf16(
            false, ahi.v, false, blo.v, (short)0, acc[mt], false, false);
        acc[mt] = __builtin_amdgcn_wmma_f32_16x16x32_bf16(
            false, alo.v, false, bhi.v, (short)0, acc[mt], false, false);
      }
    }
    __syncthreads();

    // ---------- spill scores (x 1/sqrt(64)) to Wbuf[m][t] ----------
    {
      int rowadd = (lane < 16) ? 0 : 8;
#pragma unroll
      for (int mt = 0; mt < 4; ++mt)
#pragma unroll
        for (int r = 0; r < 8; ++r)
          Wbuf[(mt << 4) + l15][(wave << 4) + r + rowadd] = acc[mt][r] * 0.125f;
    }
    __syncthreads();

    // ---------- softmax over 64 states, one thread per token ----------
    if (tid < THF) {
      float mx = -1e30f;
      for (int m = 0; m < MM; ++m) mx = fmaxf(mx, Wbuf[m][tid]);
      float sum = 0.f;
      for (int m = 0; m < MM; ++m) {
        float e = __expf(Wbuf[m][tid] - mx);
        Wbuf[m][tid] = e;
        sum += e;
      }
      float inv = 1.0f / sum;
      for (int m = 0; m < MM; ++m) Wbuf[m][tid] *= inv;
    }
    __syncthreads();

    // ---------- intra-chunk backward scan, one thread per state m ----------
    if (tid < MM) {
      float P = Pbuf[tid];
      for (int t = THF - 1; t >= 0; --t) {
        float wv = Wbuf[tid][t];
        Wbuf[tid][t] = ALPHA_C * wv * P;   // coeff[m][t]
        P *= (1.0f - ALPHA_C * wv);
      }
      Pbuf[tid] = P;
    }
    __syncthreads();

    // ---------- partial s GEMM (exact f32 WMMA): s[m][e] += coeff x Hh ----------
    {
      const int mrow  = (mt2 << 4) + l15;
      const int ecol0 = (et0 << 4) + l15;
#pragma unroll 4
      for (int kt = 0; kt < THF; kt += 4) {
        int t = kt + koff;
        v2f a = *(const v2f*)&Wbuf[mrow][t];
        const float* hb0 = Hb + (size_t)(t0 + t) * DD + hh * HD;
        v2f b0; b0.x = hb0[ecol0];       b0.y = hb0[DD + ecol0];
        v2f b1; b1.x = hb0[ecol0 + 16];  b1.y = hb0[DD + ecol0 + 16];
        sacc0 = __builtin_amdgcn_wmma_f32_16x16x4_f32(
            false, a, false, b0, (short)0, sacc0, false, false);
        sacc1 = __builtin_amdgcn_wmma_f32_16x16x4_f32(
            false, a, false, b1, (short)0, sacc1, false, false);
      }
    }
  }
  __syncthreads();

  // ---------- write s_chunk and G_chunk ----------
  {
    float* sout = s_chunks + ((((size_t)b * NH + hh) * NC + c) * MM * HD);
    int rowadd = (lane < 16) ? 0 : 8;
#pragma unroll
    for (int r = 0; r < 8; ++r) {
      int m = (mt2 << 4) + r + rowadd;
      sout[m * HD + (et0 << 4) + l15]       = sacc0[r];
      sout[m * HD + ((et0 + 1) << 4) + l15] = sacc1[r];
    }
  }
  if (tid < MM)
    G_chunks[(((size_t)b * NH + hh) * NC + c) * MM + tid] = Pbuf[tid];
}

// ---------------- Kernel 2: cross-chunk suffix combine (16 steps) ----------------
__global__ __launch_bounds__(256) void combine_kernel(
    const float* __restrict__ s_chunks, const float* __restrict__ G_chunks,
    float* __restrict__ out) {
  int bh = blockIdx.x;           // 96 blocks = B * NH
  int hh = bh % NH;
  int b  = bh / NH;
  int tid = threadIdx.x;
  int m  = tid >> 2;             // 0..63
  int e0 = (tid & 3) * 16;       // 16 e's per thread
  const float* sc = s_chunks + (((size_t)b * NH + hh) * NC) * MM * HD;
  const float* gc = G_chunks + (((size_t)b * NH + hh) * NC) * MM;
  float accv[16];
#pragma unroll
  for (int i = 0; i < 16; ++i) accv[i] = 0.f;
  float P = 1.0f;
  for (int c = NC - 1; c >= 0; --c) {
    const float* s_c = sc + (size_t)c * MM * HD + m * HD + e0;
#pragma unroll
    for (int i = 0; i < 16; ++i) accv[i] += P * s_c[i];
    P *= gc[c * MM + m];
  }
  float* o = out + ((size_t)b * MM + m) * DD + hh * HD + e0;
#pragma unroll
  for (int i = 0; i < 16; ++i) o[i] = accv[i];
}

extern "C" void kernel_launch(void* const* d_in, const int* in_sizes, int n_in,
                              void* d_out, int out_size, void* d_ws, size_t ws_size,
                              hipStream_t stream) {
  const float* H     = (const float*)d_in[0];  // [8,4096,768]
  const float* proto = (const float*)d_in[1];  // [64,768]
  const float* Wt    = (const float*)d_in[2];  // [768,768]
  // d_in[3] = s0 (zeros) -> residual term vanishes, unused

  float* ws = (float*)d_ws;
  float* KE = ws;                                  // 12*64*768
  float* G  = KE + (size_t)NH * MM * DD;           // 8*12*16*64
  float* SC = G + (size_t)BD * NH * NC * MM;       // 8*12*16*64*64  (~28 MB total)

  ke_precompute_kernel<<<(NH * MM * DD + 255) / 256, 256, 0, stream>>>(proto, Wt, KE);
  score_scan_kernel<<<BD * NH * NC, 256, 0, stream>>>(H, KE, SC, G);
  combine_kernel<<<BD * NH, 256, 0, stream>>>(SC, G, (float*)d_out);
}
